// RN_10479720202478
// MI455X (gfx1250) — matrix operand order repaired and checked by reading, not compiled, not run on hardware
//
#include <hip/hip_runtime.h>
#include <hip/hip_bf16.h>

// ---- problem sizes --------------------------------------------------------
constexpr int BB   = 128;   // batch
constexpr int NN   = 64;    // boxes
constexpr int TT   = 20;    // seq len
constexpr int ICNN = 24;
constexpr int HH   = 128;   // lstm hidden
constexpr int EE   = 300;   // embed dim
constexpr int LL   = 256;   // mlp width
constexpr int NCLS = 100;
constexpr int G4   = 4 * HH;      // 512
constexpr int DBOX = ICNN + 2;    // 26
constexpr int LPAD = LL + 8;      // bf16 LDS row stride (528B = 16B aligned)

typedef __attribute__((ext_vector_type(16))) __bf16 v16bf;
typedef __attribute__((ext_vector_type(8)))  __bf16 v8bf;
typedef __attribute__((ext_vector_type(8)))  float  v8f;

// ---- 1) tok = tanh(emb[q]) ; Xpre[b,t,:] = tok @ w_ih.T + b_ih + b_hh -----
__global__ void k_embed_xpre(const float* __restrict__ emb,
                             const int*   __restrict__ q_feats,
                             const float* __restrict__ w_ih,
                             const float* __restrict__ b_ih,
                             const float* __restrict__ b_hh,
                             float* __restrict__ xpre) {
  int bt  = blockIdx.x;            // 0..B*T-1
  int tid = threadIdx.x;           // 256
  __shared__ float tokx[EE];
  int tokid = q_feats[bt];
  for (int e = tid; e < EE; e += 256) tokx[e] = tanhf(emb[tokid * EE + e]);
  __syncthreads();
  for (int g = tid; g < G4; g += 256) {
    float acc = b_ih[g] + b_hh[g];
    const float* wr = w_ih + g * EE;
    for (int e = 0; e < EE; ++e) acc += tokx[e] * wr[e];
    xpre[bt * G4 + g] = acc;
  }
}

// ---- 2) masked LSTM recurrence, one block per batch row -------------------
__global__ void k_lstm(const float* __restrict__ xpre,
                       const float* __restrict__ w_hh,
                       const int*   __restrict__ q_lens,
                       float* __restrict__ cT) {
  int b = blockIdx.x, tid = threadIdx.x;      // 128 threads
  __shared__ float hs[HH], cs[HH];
  hs[tid] = 0.f; cs[tid] = 0.f;
  int len = q_lens[b];
  __syncthreads();
  for (int t = 0; t < TT; ++t) {
    const float* xp = xpre + (b * TT + t) * G4;
    float gi = xp[tid], gf = xp[HH + tid], gg = xp[2 * HH + tid], go = xp[3 * HH + tid];
    const float* wi = w_hh + (0 * HH + tid) * HH;
    const float* wf = w_hh + (1 * HH + tid) * HH;
    const float* wg = w_hh + (2 * HH + tid) * HH;
    const float* wo = w_hh + (3 * HH + tid) * HH;
    for (int h = 0; h < HH; ++h) {
      float hv = hs[h];
      gi += hv * wi[h]; gf += hv * wf[h]; gg += hv * wg[h]; go += hv * wo[h];
    }
    float ig = 1.f / (1.f + expf(-gi));
    float fg = 1.f / (1.f + expf(-gf));
    float og = 1.f / (1.f + expf(-go));
    float cn = fg * cs[tid] + ig * tanhf(gg);
    float hn = og * tanhf(cn);
    __syncthreads();                      // all reads of hs done
    if (t < len) { cs[tid] = cn; hs[tid] = hn; }
    __syncthreads();
  }
  cT[b * HH + tid] = cs[tid];
}

// ---- 3) P = box@W1a , Q = box@W1b  (layer-1 factorization) ----------------
__global__ void k_pq(const float* __restrict__ box_feats,
                     const float* __restrict__ g_w1,
                     float* __restrict__ P, float* __restrict__ Q) {
  int bn = blockIdx.x;                 // b*N + n
  int n  = bn & (NN - 1);
  int tid = threadIdx.x;               // 256 = LL
  __shared__ float bx[DBOX];
  if (tid < ICNN)      bx[tid]      = box_feats[bn * ICNN + tid];
  if (tid == ICNN)     bx[ICNN]     = ((n >> 3) - 4) * 0.125f;  // coord i
  if (tid == ICNN + 1) bx[ICNN + 1] = ((n & 7) - 4) * 0.125f;   // coord j
  __syncthreads();
  float accP = 0.f, accQ = 0.f;
  for (int d = 0; d < DBOX; ++d) {
    accP += bx[d] * g_w1[d * LL + tid];
    accQ += bx[d] * g_w1[(DBOX + d) * LL + tid];
  }
  P[bn * LL + tid] = accP;
  Q[bn * LL + tid] = accQ;
}

// ---- 4) R = q_rnn @ W1c + g_b1 -------------------------------------------
__global__ void k_r(const float* __restrict__ cT, const float* __restrict__ g_w1,
                    const float* __restrict__ g_b1, float* __restrict__ R) {
  int b = blockIdx.x, tid = threadIdx.x;     // 256
  __shared__ float q[HH];
  if (tid < HH) q[tid] = cT[b * HH + tid];
  __syncthreads();
  float acc = g_b1[tid];
  for (int h = 0; h < HH; ++h) acc += q[h] * g_w1[(2 * DBOX + h) * LL + tid];
  R[b * LL + tid] = acc;
}

// ---- 5) weight fp32[k][n] -> bf16 transposed [n][k] -----------------------
__global__ void k_wconv(const float* __restrict__ in, __bf16* __restrict__ out) {
  int idx = blockIdx.x * 256 + threadIdx.x;   // 65536
  int n = idx >> 8, k = idx & 255;
  out[n * LL + k] = (__bf16)in[k * LL + n];
}

__global__ void k_zero(float* p, int n) {
  int i = blockIdx.x * 256 + threadIdx.x;
  if (i < n) p[i] = 0.f;
}

// ---- 6) fused RN g-MLP: layers 2..4 via bf16 WMMA + pair-sum --------------
__global__ void __launch_bounds__(256) k_rn(
    const float* __restrict__ P, const float* __restrict__ Q, const float* __restrict__ R,
    const __bf16* __restrict__ Wt2, const float* __restrict__ b2,
    const __bf16* __restrict__ Wt3, const float* __restrict__ b3,
    const __bf16* __restrict__ Wt4, const float* __restrict__ b4,
    float* __restrict__ g_out) {
  int blk = blockIdx.x;           // b*N + i
  int b   = blk >> 6;
  int tid = threadIdx.x;

  __shared__ __align__(16) __bf16 xa[64 * LPAD];
  __shared__ __align__(16) __bf16 xb[64 * LPAD];
  __shared__ float psum[LL];

  // build x1[j, c] = relu(P[b,i,c] + Q[b,j,c] + R[b,c]) as bf16
  {
    float pr = P[blk * LL + tid] + R[b * LL + tid];
    const float* qb = Q + (size_t)b * NN * LL + tid;
    for (int j = 0; j < 64; ++j) {
      float v = pr + qb[j * LL];
      xa[j * LPAD + tid] = (__bf16)fmaxf(v, 0.f);
    }
    psum[tid] = 0.f;
  }
  __syncthreads();

  const int lane   = tid & 31;
  const int wave   = tid >> 5;
  const int mt     = wave & 3;              // M-tile (16 rows each)
  const int ntbase = (wave >> 2) * 8;       // N-tile half
  const int m_lo   = lane & 15;
  const int hi     = lane >> 4;             // 0 | 1
  const int khA    = hi ? 8 : 0;            // A: K-sub per ISA 16-bit A layout
  const int khB    = hi ? 8 : 0;            // B: same K-sub layout as A
  const int rowA   = mt * 16 + m_lo;

  for (int layer = 0; layer < 3; ++layer) {
    // select without pointer arrays (avoids LDS addrspacecast static init)
    const __bf16* xin  = (layer & 1) ? xb : xa;
    __bf16*       xout = (layer & 1) ? xa : xb;
    const __bf16* Wt   = (layer == 0) ? Wt2 : ((layer == 1) ? Wt3 : Wt4);
    const float*  bias = (layer == 0) ? b2  : ((layer == 1) ? b3  : b4);
    __builtin_prefetch(Wt + (ntbase * 16 + m_lo) * LL, 0, 1);

    for (int nt = ntbase; nt < ntbase + 8; ++nt) {
      int col = nt * 16 + m_lo;
      v8f acc = {};
#pragma unroll
      for (int kk = 0; kk < 8; ++kk) {
        int kbase = kk * 32;
        union { v16bf v; v8bf h[2]; } A, Bv;
        const __bf16* ap = xin + rowA * LPAD + kbase + khA;
        A.h[0] = *(const v8bf*)(ap);
        A.h[1] = *(const v8bf*)(ap + 16);
        const __bf16* wp = Wt + col * LL + kbase + khB;
        Bv.h[0] = *(const v8bf*)(wp);
        Bv.h[1] = *(const v8bf*)(wp + 16);
        acc = __builtin_amdgcn_wmma_f32_16x16x32_bf16(
            false, A.v, false, Bv.v, (short)0, acc, false, false);
      }
      float bc = bias[col];
      if (layer < 2) {
        // D layout: VGPR r -> M = r + 8*hi, N = lane&15
        int mrow = mt * 16 + hi * 8;
#pragma unroll
        for (int r = 0; r < 8; ++r) {
          float v = fmaxf(acc[r] + bc, 0.f);
          xout[(mrow + r) * LPAD + col] = (__bf16)v;
        }
      } else {
        float s = 0.f;
#pragma unroll
        for (int r = 0; r < 8; ++r) s += fmaxf(acc[r] + bc, 0.f);
        atomicAdd(&psum[col], s);          // ds_add_f32
      }
    }
    __syncthreads();                        // layer boundary
  }

  atomicAdd(&g_out[b * LL + tid], psum[tid]);  // global_atomic_add_f32
}

// ---- 7) f-MLP: 256->256->256->100 -----------------------------------------
__global__ void k_f(const float* __restrict__ g_out,
                    const float* __restrict__ f_w1, const float* __restrict__ f_b1,
                    const float* __restrict__ f_w2, const float* __restrict__ f_b2,
                    const float* __restrict__ f_w3, const float* __restrict__ f_b3,
                    float* __restrict__ out) {
  int b = blockIdx.x, tid = threadIdx.x;   // 256
  __shared__ float x0[LL], x1[LL], x2[LL];
  x0[tid] = g_out[b * LL + tid];
  __syncthreads();
  float a = f_b1[tid];
  for (int k = 0; k < LL; ++k) a += x0[k] * f_w1[k * LL + tid];
  x1[tid] = fmaxf(a, 0.f);
  __syncthreads();
  a = f_b2[tid];
  for (int k = 0; k < LL; ++k) a += x1[k] * f_w2[k * LL + tid];
  x2[tid] = fmaxf(a, 0.f);
  __syncthreads();
  if (tid < NCLS) {
    a = f_b3[tid];
    for (int k = 0; k < LL; ++k) a += x2[k] * f_w3[k * NCLS + tid];
    out[b * NCLS + tid] = a;
  }
}

// ---- host launch ----------------------------------------------------------
extern "C" void kernel_launch(void* const* d_in, const int* in_sizes, int n_in,
                              void* d_out, int out_size, void* d_ws, size_t ws_size,
                              hipStream_t stream) {
  const float* box_feats = (const float*)d_in[0];
  const float* emb  = (const float*)d_in[1];
  const float* w_ih = (const float*)d_in[2];
  const float* w_hh = (const float*)d_in[3];
  const float* b_ih = (const float*)d_in[4];
  const float* b_hh = (const float*)d_in[5];
  const float* g_w1 = (const float*)d_in[6];
  const float* g_b1 = (const float*)d_in[7];
  const float* g_w2 = (const float*)d_in[8];
  const float* g_b2 = (const float*)d_in[9];
  const float* g_w3 = (const float*)d_in[10];
  const float* g_b3 = (const float*)d_in[11];
  const float* g_w4 = (const float*)d_in[12];
  const float* g_b4 = (const float*)d_in[13];
  const float* f_w1 = (const float*)d_in[14];
  const float* f_b1 = (const float*)d_in[15];
  const float* f_w2 = (const float*)d_in[16];
  const float* f_b2 = (const float*)d_in[17];
  const float* f_w3 = (const float*)d_in[18];
  const float* f_b3 = (const float*)d_in[19];
  const int* q_feats = (const int*)d_in[20];
  const int* q_lens  = (const int*)d_in[21];
  float* out = (float*)d_out;

  // workspace carve-up (~23 MB)
  float* ws_f = (float*)d_ws;
  float* xpre = ws_f;                        // 128*20*512
  float* cT   = xpre + (size_t)BB * TT * G4; // 128*128
  float* Pp   = cT + BB * HH;                // 128*64*256
  float* Qp   = Pp + (size_t)BB * NN * LL;
  float* Rp   = Qp + (size_t)BB * NN * LL;   // 128*256
  float* gout = Rp + BB * LL;                // 128*256
  __bf16* Wt2 = (__bf16*)(gout + BB * LL);   // 3 x 256*256 bf16
  __bf16* Wt3 = Wt2 + LL * LL;
  __bf16* Wt4 = Wt3 + LL * LL;

  k_embed_xpre<<<BB * TT, 256, 0, stream>>>(emb, q_feats, w_ih, b_ih, b_hh, xpre);
  k_lstm<<<BB, HH, 0, stream>>>(xpre, w_hh, q_lens, cT);
  k_pq<<<BB * NN, 256, 0, stream>>>(box_feats, g_w1, Pp, Qp);
  k_r<<<BB, 256, 0, stream>>>(cT, g_w1, g_b1, Rp);
  k_wconv<<<256, 256, 0, stream>>>(g_w2, Wt2);
  k_wconv<<<256, 256, 0, stream>>>(g_w3, Wt3);
  k_wconv<<<256, 256, 0, stream>>>(g_w4, Wt4);
  k_zero<<<128, 256, 0, stream>>>(gout, BB * LL);
  k_rn<<<BB * NN, 256, 0, stream>>>(Pp, Qp, Rp, Wt2, g_b2, Wt3, g_b3, Wt4, g_b4, gout);
  k_f<<<BB, 256, 0, stream>>>(gout, f_w1, f_b1, f_w2, f_b2, f_w3, f_b3, out);
}
